// SharedSDFDecoder_20169166422585
// MI455X (gfx1250) — matrix operand order repaired and checked by reading, not compiled, not run on hardware
//
#include <hip/hip_runtime.h>

// ---------------------------------------------------------------------------
// SharedSDFDecoder for MI455X (gfx1250, wave32, WMMA bf16)
//   points    [8, 200000, 3] f32     triplanes [8, 3, 32, 256, 256] f32
//   W1 [99,256] b1[256] W2[256,256] b2[256] W3[256,1] b3[1]
//   out       [8, 200000, 1] f32
// ---------------------------------------------------------------------------

typedef __attribute__((ext_vector_type(16))) __bf16 v16bf;
typedef __attribute__((ext_vector_type(8)))  __bf16 v8bf;
typedef __attribute__((ext_vector_type(4)))  __bf16 v4bf;
typedef __attribute__((ext_vector_type(8)))  float  v8f;
typedef __attribute__((ext_vector_type(4)))  float  v4f;

#define TP_N       200000
#define PLANE_ELEM 65536          // 256*256
#define W1_FRAGS   (16*4)         // 16 n-tiles x 4 k-steps (K padded 99->128)
#define W2_FRAGS   (16*8)         // 16 n-tiles x 8 k-steps (K=256)
#define FRAG_HALFS 512            // 32 lanes x 16 halfs
#define FRAG_BYTES ((W1_FRAGS + W2_FRAGS) * FRAG_HALFS * 2)   // 192 KB
#define TPOSE_ELEMS ((size_t)24 * PLANE_ELEM * 32)            // 50,331,648
#define TPOSE_BYTES (TPOSE_ELEMS * 4)                         // ~201 MB

__device__ __forceinline__ __bf16 f2bf(float f) { return (__bf16)f; }

// single-instruction ReLU: v_med3_f32(x, 0, +inf)
__device__ __forceinline__ float relu(float x) {
  return __builtin_amdgcn_fmed3f(x, 0.0f, __builtin_inff());
}

// ---------------------------------------------------------------------------
// Weight prep: f32 row-major [K][256] -> bf16 WMMA B-fragments.
// B layout (32x16 bf16, wave32): lane holds column N = 16*nt + (lane&15),
// halfs e=0..15 hold K = 32*ks + 8*(lane>>4) + (e&7) + 16*(e>>3).  Stored [frag][lane][e].
// ---------------------------------------------------------------------------
__global__ void prep_w_kernel(const float* __restrict__ W, __bf16* __restrict__ dst,
                              int Ksrc, int nks, int total) {
  int idx = blockIdx.x * blockDim.x + threadIdx.x;
  if (idx >= total) return;
  int f  = idx >> 9;
  int r  = idx & 511;
  int ln = r >> 4;
  int e  = r & 15;
  int nt = f / nks;
  int ks = f - nt * nks;
  int K  = 32 * ks + 8 * (ln >> 4) + (e & 7) + 16 * (e >> 3);
  int n  = 16 * nt + (ln & 15);
  float v = (K < Ksrc) ? W[(size_t)K * 256 + n] : 0.0f;
  dst[idx] = f2bf(v);
}

// Channel-last transpose: src [24][32][65536] -> dst [24][65536][32] (f32).
// Lanes sweep c (32 distinct lines); successive pix iterations reuse those
// lines from L0/L2, writes fully coalesced.
__global__ void transpose_tp_kernel(const float* __restrict__ src,
                                    float* __restrict__ dst) {
  size_t idx = (size_t)blockIdx.x * 256 + threadIdx.x;
  int c       = (int)(idx & 31);
  size_t rest = idx >> 5;
  int pix     = (int)(rest & (PLANE_ELEM - 1));
  int bp      = (int)(rest >> 16);
  dst[idx] = src[((size_t)bp * 32 + c) * PLANE_ELEM + pix];
}

struct BilinW { int o00, o01, o10, o11; float wx, wy; };

__device__ __forceinline__ BilinW bilin_setup(float cx, float cy) {
  float fx = fminf(fmaxf((cx + 1.0f) * 0.5f * 255.0f, 0.0f), 255.0f);
  float fy = fminf(fmaxf((cy + 1.0f) * 0.5f * 255.0f, 0.0f), 255.0f);
  float x0f = floorf(fx), y0f = floorf(fy);
  int ix0 = (int)x0f, iy0 = (int)y0f;
  int ix1 = ix0 + 1 < 255 ? ix0 + 1 : 255;
  int iy1 = iy0 + 1 < 255 ? iy0 + 1 : 255;
  BilinW w;
  w.wx = fx - x0f; w.wy = fy - y0f;
  w.o00 = iy0 * 256 + ix0; w.o01 = iy0 * 256 + ix1;
  w.o10 = iy1 * 256 + ix0; w.o11 = iy1 * 256 + ix1;
  return w;
}

// FAST: channel-last plane [65536][32]; one corner = contiguous 128B line,
// consumed as float4 vector loads; results packed to ds_store_b64.
__device__ __forceinline__ void sample_fast(const float* __restrict__ pl,
                                            float cx, float cy, int clo, int chi,
                                            __bf16* dstRow) {
  BilinW w = bilin_setup(cx, cy);
  const float* p00 = pl + (size_t)w.o00 * 32;
  const float* p01 = pl + (size_t)w.o01 * 32;
  const float* p10 = pl + (size_t)w.o10 * 32;
  const float* p11 = pl + (size_t)w.o11 * 32;
#pragma unroll
  for (int c = clo; c < chi; c += 4) {
    v4f v00 = *(const v4f*)(p00 + c);
    v4f v01 = *(const v4f*)(p01 + c);
    v4f v10 = *(const v4f*)(p10 + c);
    v4f v11 = *(const v4f*)(p11 + c);
    v4f top = v00 + w.wx * (v01 - v00);
    v4f bot = v10 + w.wx * (v11 - v10);
    v4f v   = top + w.wy * (bot - top);
    v4bf pk;
#pragma unroll
    for (int i = 0; i < 4; ++i) pk[i] = f2bf(v[i]);
    *(v4bf*)(dstRow + c) = pk;
  }
}

// SLOW fallback: original channel-strided f32 plane [32][256][256].
__device__ __forceinline__ void sample_slow(const float* __restrict__ pl,
                                            float cx, float cy, int clo, int chi,
                                            __bf16* dstRow) {
  BilinW w = bilin_setup(cx, cy);
#pragma unroll 4
  for (int c = clo; c < chi; ++c) {
    const float* ch = pl + (size_t)c * PLANE_ELEM;
    float v00 = ch[w.o00], v01 = ch[w.o01], v10 = ch[w.o10], v11 = ch[w.o11];
    float top = v00 + w.wx * (v01 - v00);
    float bot = v10 + w.wx * (v11 - v10);
    dstRow[c] = f2bf(top + w.wy * (bot - top));
  }
}

// A-fragment from an LDS bf16 row (16-bit A layout: lane holds row M=lane&15;
// two contiguous 8-half chunks at K = 32*ks + 8*(lane>>4) and +16).
__device__ __forceinline__ v16bf load_a_frag(const __bf16* rowBase, int ks, int h) {
  const v8bf lo = *(const v8bf*)(rowBase + 32 * ks + 8 * h);
  const v8bf hi = *(const v8bf*)(rowBase + 32 * ks + 16 + 8 * h);
  return __builtin_shufflevector(lo, hi, 0, 1, 2, 3, 4, 5, 6, 7,
                                         8, 9, 10, 11, 12, 13, 14, 15);
}

// 64 threads = 2 waves; each wave owns 32 points as two 16-row M-subtiles so
// every B-fragment load feeds two WMMAs.
template <bool FAST>
__global__ __launch_bounds__(64)
void sdf_main_kernel(const float* __restrict__ pts,
                     const float* __restrict__ planes,   // FAST: chan-last ws; else triplanes
                     const __bf16* __restrict__ w1f,
                     const float* __restrict__ b1,
                     const __bf16* __restrict__ w2f,
                     const float* __restrict__ b2,
                     const float* __restrict__ w3,
                     const float* __restrict__ b3,
                     float* __restrict__ out) {
  __shared__ __align__(16) __bf16 s_feats[2][2][16][136];  // [wave][tile][row][K]
  __shared__ __align__(16) __bf16 s_h1[2][2][16][264];

  const int tid  = threadIdx.x;
  const int wave = tid >> 5;
  const int lane = tid & 31;
  const int m    = lane & 15;
  const int h    = lane >> 4;

  const long long base = (long long)blockIdx.x * 64 + (long long)wave * 32;
  const int b = (int)(base / TP_N);   // wave-uniform (tiles are 16-aligned in N)
  const size_t planeStride = (size_t)32 * PLANE_ELEM;
  const float* tb = planes + (size_t)b * 3 * planeStride;

  // ---- sampling: each lane fills half the 96 channels for 2 points --------
#pragma unroll
  for (int t = 0; t < 2; ++t) {
    const long long pt = base + 16 * t + m;
    const float p0 = pts[pt * 3 + 0];
    const float p1 = pts[pt * 3 + 1];
    const float p2 = pts[pt * 3 + 2];
    __bf16* row = &s_feats[wave][t][m][0];
    if (FAST) {
      if (h == 0) {
        sample_fast(tb,                   p0, p1, 0, 32, row);        // xy
        sample_fast(tb + planeStride,     p1, p2, 0, 16, row + 32);   // yz lo
      } else {
        sample_fast(tb + planeStride,     p1, p2, 16, 32, row + 32);  // yz hi
        sample_fast(tb + 2 * planeStride, p0, p2, 0, 32, row + 64);   // xz
      }
    } else {
      if (h == 0) {
        sample_slow(tb,                   p0, p1, 0, 32, row);
        sample_slow(tb + planeStride,     p1, p2, 0, 16, row + 32);
      } else {
        sample_slow(tb + planeStride,     p1, p2, 16, 32, row + 32);
        sample_slow(tb + 2 * planeStride, p0, p2, 0, 32, row + 64);
      }
    }
    if (h == 0) {
      row[96] = f2bf(p0); row[97] = f2bf(p1); row[98] = f2bf(p2);
#pragma unroll
      for (int k = 99; k < 112; ++k) row[k] = f2bf(0.0f);
    } else {
#pragma unroll
      for (int k = 112; k < 128; ++k) row[k] = f2bf(0.0f);
    }
  }

  __syncthreads();

  // ---- layer 1: h1 = relu(feats @ W1 + b1), K=128, N=256 ------------------
  for (int nt = 0; nt < 16; ++nt) {
    const float bias = b1[16 * nt + m];
    v8f acc0, acc1;
#pragma unroll
    for (int r = 0; r < 8; ++r) { acc0[r] = bias; acc1[r] = bias; }
#pragma unroll
    for (int ks = 0; ks < 4; ++ks) {
      v16bf bb = *(const v16bf*)(w1f + (size_t)(nt * 4 + ks) * FRAG_HALFS + lane * 16);
      v16bf a0 = load_a_frag(&s_feats[wave][0][m][0], ks, h);
      v16bf a1 = load_a_frag(&s_feats[wave][1][m][0], ks, h);
      acc0 = __builtin_amdgcn_wmma_f32_16x16x32_bf16(false, a0, false, bb, (short)0, acc0, false, false);
      acc1 = __builtin_amdgcn_wmma_f32_16x16x32_bf16(false, a1, false, bb, (short)0, acc1, false, false);
    }
#pragma unroll
    for (int r = 0; r < 8; ++r) {
      s_h1[wave][0][8 * h + r][16 * nt + m] = f2bf(relu(acc0[r]));
      s_h1[wave][1][8 * h + r][16 * nt + m] = f2bf(relu(acc1[r]));
    }
  }

  __syncthreads();

  // ---- layer 2 + fused layer 3 --------------------------------------------
  float acc3[2][8];
#pragma unroll
  for (int t = 0; t < 2; ++t)
#pragma unroll
    for (int r = 0; r < 8; ++r) acc3[t][r] = 0.0f;

  for (int nt = 0; nt < 16; ++nt) {
    const float bias = b2[16 * nt + m];
    const float w3v  = w3[16 * nt + m];
    v8f acc0, acc1;
#pragma unroll
    for (int r = 0; r < 8; ++r) { acc0[r] = bias; acc1[r] = bias; }
#pragma unroll
    for (int ks = 0; ks < 8; ++ks) {
      v16bf bb = *(const v16bf*)(w2f + (size_t)(nt * 8 + ks) * FRAG_HALFS + lane * 16);
      v16bf a0 = load_a_frag(&s_h1[wave][0][m][0], ks, h);
      v16bf a1 = load_a_frag(&s_h1[wave][1][m][0], ks, h);
      acc0 = __builtin_amdgcn_wmma_f32_16x16x32_bf16(false, a0, false, bb, (short)0, acc0, false, false);
      acc1 = __builtin_amdgcn_wmma_f32_16x16x32_bf16(false, a1, false, bb, (short)0, acc1, false, false);
    }
#pragma unroll
    for (int r = 0; r < 8; ++r) {
      acc3[0][r] += relu(acc0[r]) * w3v;
      acc3[1][r] += relu(acc1[r]) * w3v;
    }
  }

  // reduce across the 16 column residues (stays within each 16-lane half)
#pragma unroll
  for (int o = 1; o < 16; o <<= 1)
#pragma unroll
    for (int t = 0; t < 2; ++t)
#pragma unroll
      for (int r = 0; r < 8; ++r)
        acc3[t][r] += __shfl_xor(acc3[t][r], o, 32);

  const float b3s = b3[0];
  if (m == 0) {
#pragma unroll
    for (int t = 0; t < 2; ++t)
#pragma unroll
      for (int r = 0; r < 8; ++r)
        out[base + 16 * t + 8 * h + r] = acc3[t][r] + b3s;
  }
}

extern "C" void kernel_launch(void* const* d_in, const int* in_sizes, int n_in,
                              void* d_out, int out_size, void* d_ws, size_t ws_size,
                              hipStream_t stream) {
  const float* pts = (const float*)d_in[0];
  const float* tp  = (const float*)d_in[1];
  const float* W1  = (const float*)d_in[2];
  const float* b1  = (const float*)d_in[3];
  const float* W2  = (const float*)d_in[4];
  const float* b2  = (const float*)d_in[5];
  const float* W3  = (const float*)d_in[6];
  const float* b3  = (const float*)d_in[7];
  float* out = (float*)d_out;

  __bf16* w1f = (__bf16*)d_ws;                        // 64 KB of fragments
  __bf16* w2f = w1f + W1_FRAGS * FRAG_HALFS;          // 128 KB of fragments

  const int t1 = W1_FRAGS * FRAG_HALFS;   // 32768
  const int t2 = W2_FRAGS * FRAG_HALFS;   // 65536
  prep_w_kernel<<<(t1 + 255) / 256, 256, 0, stream>>>(W1, w1f, 99, 4, t1);
  prep_w_kernel<<<(t2 + 255) / 256, 256, 0, stream>>>(W2, w2f, 256, 8, t2);

  const bool fast = ws_size >= (size_t)FRAG_BYTES + TPOSE_BYTES;
  if (fast) {
    float* tpose = (float*)((char*)d_ws + FRAG_BYTES);   // 128B-aligned
    transpose_tp_kernel<<<(unsigned)(TPOSE_ELEMS / 256), 256, 0, stream>>>(tp, tpose);
    sdf_main_kernel<true><<<25000, 64, 0, stream>>>(pts, tpose, w1f, b1, w2f, b2, W3, b3, out);
  } else {
    sdf_main_kernel<false><<<25000, 64, 0, stream>>>(pts, tp, w1f, b1, w2f, b2, W3, b3, out);
  }
}
